// Mamba_25855703121921
// MI455X (gfx1250) — matrix operand order repaired and checked by reading, not compiled, not run on hardware
//
#include <hip/hip_runtime.h>
#include <hip/hip_bf16.h>

// ---------------------------------------------------------------------------
// Mamba forward for MI455X (gfx1250, wave32).
// GEMMs: v_wmma_f32_16x16x32_bf16 (fp32 accumulate), 128x128x32 tiles,
// double-buffered LDS fed by async global->LDS copies (ASYNCcnt) when the
// toolchain exposes the gfx1250 builtins. Conv / softplus / scan / gating in
// fp32 VALU.
// ---------------------------------------------------------------------------

#define B_SZ     2
#define SEQ_L    2048
#define D_MODEL  1024
#define D_INNER  2048
#define D_STATE  16
#define D_CONV   4
#define DT_RANK  64
#define M_TOK    (B_SZ * SEQ_L)          // 4096 token rows
#define XDBL_LD  128                     // padded row stride for x_dbl (>=96)

typedef __bf16 bf16_t;
typedef __attribute__((ext_vector_type(16))) __bf16 v16bf;
typedef __attribute__((ext_vector_type(8)))  __bf16 v8bf;
typedef __attribute__((ext_vector_type(8)))  float  v8f;
typedef __attribute__((ext_vector_type(4)))  int    v4i;

#ifndef __has_builtin
#define __has_builtin(x) 0
#endif

#if __has_builtin(__builtin_amdgcn_global_load_async_to_lds_b128)
#define HAVE_ASYNC 1
#else
#define HAVE_ASYNC 0
#endif

#if HAVE_ASYNC
// builtin signature (from clang diagnostic): (v4i AS1*, v4i AS3*, Imm, Imm)
typedef __attribute__((address_space(1))) v4i as1_v4i;
typedef __attribute__((address_space(3))) v4i as3_v4i;

__device__ __forceinline__ void async_cp16(const void* g, void* l) {
    // 16B global -> LDS, tracked by ASYNCcnt, no VGPR staging
    __builtin_amdgcn_global_load_async_to_lds_b128(
        (as1_v4i*)(void*)g, (as3_v4i*)l, 0, 0);
}

__device__ __forceinline__ void wait_async0() {
#if __has_builtin(__builtin_amdgcn_s_wait_asynccnt)
    __builtin_amdgcn_s_wait_asynccnt(0);
#else
    asm volatile("s_wait_asynccnt 0x0" ::: "memory");
#endif
}
#endif

#define SHUF16(lo, hi) __builtin_shufflevector((lo), (hi), \
    0, 1, 2, 3, 4, 5, 6, 7, 8, 9, 10, 11, 12, 13, 14, 15)

// ---------------------------------------------------------------------------
// fp32 -> bf16 convert (same layout)
// ---------------------------------------------------------------------------
__global__ void k_convert_bf16(const float* __restrict__ in,
                               bf16_t* __restrict__ out, int n) {
    int i = blockIdx.x * blockDim.x + threadIdx.x;
    if (i < n) out[i] = (bf16_t)in[i];
}

// ---------------------------------------------------------------------------
// fp32 [K][N] -> bf16 [N][K] (transpose + convert); coalesced read
// ---------------------------------------------------------------------------
__global__ void k_transpose_bf16(const float* __restrict__ in,
                                 bf16_t* __restrict__ out, int K, int N) {
    int i = blockIdx.x * blockDim.x + threadIdx.x;
    if (i >= K * N) return;
    int k = i / N;
    int n = i - k * N;
    out[(size_t)n * K + k] = (bf16_t)in[i];
}

// ---------------------------------------------------------------------------
// BF16 WMMA GEMM, compile-time shapes (all multiples of the tile -> no guards)
//   C[M][N] (fp32) = A[M][K] (bf16) * Bt[N][K] (bf16, pre-transposed)
// Block = 256 threads (8 waves). Tile 128x128, K-step 32.
// Wave w owns rows [w*16, w*16+16), all 128 cols: 8 accumulators, 8 wmma/iter.
// ---------------------------------------------------------------------------
#define BM 128
#define BN 128
#define BK 32
#define LDS_PAD 8                 // row stride 80B: every 16B chunk aligned
#define LDSW (BK + LDS_PAD)

template<int M, int N, int K>
__global__ __launch_bounds__(256)
void k_wmma_gemm(const bf16_t* __restrict__ A,
                 const bf16_t* __restrict__ Bt,
                 float* __restrict__ C) {
    __shared__ bf16_t As[2][BM][LDSW];
    __shared__ bf16_t Bs[2][BN][LDSW];

    const int tid  = threadIdx.x;
    const int wave = tid >> 5;
    const int lane = tid & 31;
    const int r    = lane & 15;
    const int hsel = lane >> 4;
    const int m0   = blockIdx.y * BM;
    const int n0   = blockIdx.x * BN;

    const v8f fz = {0.f, 0.f, 0.f, 0.f, 0.f, 0.f, 0.f, 0.f};
    v8f acc[8];
#pragma unroll
    for (int j = 0; j < 8; ++j) acc[j] = fz;

    // fragment compute over one LDS buffer: 1 A-frag reused by 8 wmma
    auto compute = [&](int buf) {
        const int kbA = hsel * 8;   // lane<16: K {0..7}u{16..23}; else {8..15}u{24..31}
        const bf16_t* arow = &As[buf][wave * 16 + r][0];
        v8bf alo = *(const v8bf*)(arow + kbA);
        v8bf ahi = *(const v8bf*)(arow + kbA + 16);
        v16bf af = SHUF16(alo, ahi);
        const int kbB = hsel * 8;   // lane<16: K {0..7}u{16..23}; else {8..15}u{24..31}
#pragma unroll
        for (int j = 0; j < 8; ++j) {
            const bf16_t* brow = &Bs[buf][j * 16 + r][0];
            v8bf blo = *(const v8bf*)(brow + kbB);
            v8bf bhi = *(const v8bf*)(brow + kbB + 16);
            v16bf bfr = SHUF16(blo, bhi);
            acc[j] = __builtin_amdgcn_wmma_f32_16x16x32_bf16(
                         false, af, false, bfr, (short)0, acc[j], false, false);
        }
    };

#if HAVE_ASYNC
    // 128x32 bf16 tile = 512 x 16B chunks; 256 threads -> 2 chunks each (A and B)
    auto stage_async = [&](int buf, int k0) {
#pragma unroll
        for (int i = 0; i < 2; ++i) {
            int c   = i * 256 + tid;
            int row = c >> 2;
            int cch = c & 3;
            async_cp16(A  + (size_t)(m0 + row) * K + k0 + cch * 8,
                       &As[buf][row][cch * 8]);
            async_cp16(Bt + (size_t)(n0 + row) * K + k0 + cch * 8,
                       &Bs[buf][row][cch * 8]);
        }
    };

    int cur = 0;
    stage_async(0, 0);
    for (int k0 = 0; k0 < K; k0 += BK) {
        wait_async0();          // my async copies for buf[cur] have landed
        __syncthreads();        // ...and everyone else's too
        if (k0 + BK < K) stage_async(cur ^ 1, k0 + BK);  // overlap with wmma
        compute(cur);
        __syncthreads();        // all reads of buf[cur] done before its reuse
        cur ^= 1;
    }
#else
    for (int k0 = 0; k0 < K; k0 += BK) {
        v8bf va[2], vb[2];
#pragma unroll
        for (int i = 0; i < 2; ++i) {
            int c   = i * 256 + tid;
            int row = c >> 2;
            int cch = c & 3;
            va[i] = *(const v8bf*)(A  + (size_t)(m0 + row) * K + k0 + cch * 8);
            vb[i] = *(const v8bf*)(Bt + (size_t)(n0 + row) * K + k0 + cch * 8);
        }
        if (k0 + BK < K) {
            int pr = (tid >> 1);
            __builtin_prefetch(A  + (size_t)(m0 + pr) * K + k0 + BK, 0, 1);
            __builtin_prefetch(Bt + (size_t)(n0 + pr) * K + k0 + BK, 0, 1);
        }
        __syncthreads();
#pragma unroll
        for (int i = 0; i < 2; ++i) {
            int c   = i * 256 + tid;
            int row = c >> 2;
            int cch = c & 3;
            *(v8bf*)&As[0][row][cch * 8] = va[i];
            *(v8bf*)&Bs[0][row][cch * 8] = vb[i];
        }
        __syncthreads();
        compute(0);
    }
#endif

    // epilogue: VGPR i of acc holds (M = wave*16 + hsel*8 + i, N = j*16 + (lane&15))
#pragma unroll
    for (int j = 0; j < 8; ++j) {
        int gc = n0 + j * 16 + r;
        float* cp = C + (size_t)(m0 + wave * 16 + hsel * 8) * N + gc;
#pragma unroll
        for (int i = 0; i < 8; ++i) cp[(size_t)i * N] = acc[j][i];
    }
}

// ---------------------------------------------------------------------------
// Causal depthwise conv1d (width 4) + bias + SiLU.  xz is [M_TOK][2*D_INNER];
// xs is cols [0,D_INNER).  Writes u in fp32 and bf16.
// ---------------------------------------------------------------------------
__global__ void k_conv_silu(const float* __restrict__ xz,
                            const float* __restrict__ conv_w,
                            const float* __restrict__ conv_b,
                            float* __restrict__ u_f32,
                            bf16_t* __restrict__ u_bf) {
    int idx = blockIdx.x * blockDim.x + threadIdx.x;
    if (idx >= M_TOK * D_INNER) return;
    int t = idx / D_INNER;
    int d = idx - t * D_INNER;
    int b = t / SEQ_L;
    int l = t - b * SEQ_L;

    float acc = conv_b[d];
#pragma unroll
    for (int j = 0; j < D_CONV; ++j) {
        int ll = l - (D_CONV - 1) + j;
        if (ll >= 0)
            acc += xz[((size_t)(b * SEQ_L + ll)) * (2 * D_INNER) + d] * conv_w[d * D_CONV + j];
    }
    float s = acc / (1.f + __expf(-acc));   // silu
    u_f32[idx] = s;
    u_bf[idx]  = (bf16_t)s;
}

// ---------------------------------------------------------------------------
// Slice dt columns of x_dbl [M_TOK][XDBL_LD] -> bf16 [M_TOK][64]
// ---------------------------------------------------------------------------
__global__ void k_slice_dt(const float* __restrict__ x_dbl,
                           bf16_t* __restrict__ dt_bf) {
    int idx = blockIdx.x * blockDim.x + threadIdx.x;
    if (idx >= M_TOK * DT_RANK) return;
    int t = idx / DT_RANK;
    int r = idx - t * DT_RANK;
    dt_bf[idx] = (bf16_t)x_dbl[(size_t)t * XDBL_LD + r];
}

// ---------------------------------------------------------------------------
// Selective scan: one thread per (b, d_inner) channel, h[16] in registers,
// sequential over L.  softplus fused on the delta load.
// ---------------------------------------------------------------------------
__global__ __launch_bounds__(256)
void k_scan(const float* __restrict__ predelta,   // [M_TOK][D_INNER]
            const float* __restrict__ b_dt,       // [D_INNER]
            const float* __restrict__ x_dbl,      // [M_TOK][XDBL_LD]
            const float* __restrict__ u_f32,      // [M_TOK][D_INNER]
            const float* __restrict__ A_log,      // [D_INNER][16]
            float* __restrict__ ys) {             // [M_TOK][D_INNER]
    int idx = blockIdx.x * blockDim.x + threadIdx.x;
    if (idx >= B_SZ * D_INNER) return;
    int b = idx / D_INNER;
    int d = idx - b * D_INNER;

    float Arow[D_STATE];
#pragma unroll
    for (int n = 0; n < D_STATE; ++n) Arow[n] = -__expf(A_log[d * D_STATE + n]);

    float h[D_STATE];
#pragma unroll
    for (int n = 0; n < D_STATE; ++n) h[n] = 0.f;

    const float bias = b_dt[d];

    for (int l = 0; l < SEQ_L; ++l) {
        int t = b * SEQ_L + l;
        float pre = predelta[(size_t)t * D_INNER + d] + bias;
        float delta = (pre > 20.f) ? pre : log1pf(__expf(pre));
        float uu = u_f32[(size_t)t * D_INNER + d];

        const float*  xr  = x_dbl + (size_t)t * XDBL_LD;
        const float4* Bv4 = (const float4*)(xr + DT_RANK);            // cols 64..79
        const float4* Cv4 = (const float4*)(xr + DT_RANK + D_STATE);  // cols 80..95

        float y = 0.f;
#pragma unroll
        for (int q = 0; q < 4; ++q) {
            float4 Bq = Bv4[q];
            float4 Cq = Cv4[q];
            float bb[4] = {Bq.x, Bq.y, Bq.z, Bq.w};
            float cc[4] = {Cq.x, Cq.y, Cq.z, Cq.w};
#pragma unroll
            for (int s = 0; s < 4; ++s) {
                int n = q * 4 + s;
                float dA = __expf(delta * Arow[n]);
                h[n] = dA * h[n] + delta * bb[s] * uu;
                y += h[n] * cc[s];
            }
        }
        ys[(size_t)t * D_INNER + d] = y;
    }
}

// ---------------------------------------------------------------------------
// y = (ys + u*D) * silu(res);  res = xz cols [D_INNER, 2*D_INNER)
// ---------------------------------------------------------------------------
__global__ void k_gate(const float* __restrict__ ys,
                       const float* __restrict__ u_f32,
                       const float* __restrict__ Dvec,
                       const float* __restrict__ xz,
                       bf16_t* __restrict__ y_bf) {
    int idx = blockIdx.x * blockDim.x + threadIdx.x;
    if (idx >= M_TOK * D_INNER) return;
    int t = idx / D_INNER;
    int d = idx - t * D_INNER;
    float y   = ys[idx] + u_f32[idx] * Dvec[d];
    float res = xz[(size_t)t * (2 * D_INNER) + D_INNER + d];
    float g   = res / (1.f + __expf(-res));
    y_bf[idx] = (bf16_t)(y * g);
}

// ---------------------------------------------------------------------------
// Host-side orchestration
// ---------------------------------------------------------------------------
static inline size_t align256(size_t x) { return (x + 255) & ~(size_t)255; }

extern "C" void kernel_launch(void* const* d_in, const int* in_sizes, int n_in,
                              void* d_out, int out_size, void* d_ws, size_t ws_size,
                              hipStream_t stream) {
    const float* x      = (const float*)d_in[0];   // [2,2048,1024]
    const float* W_in   = (const float*)d_in[1];   // [1024,4096]
    const float* conv_w = (const float*)d_in[2];   // [2048,4]
    const float* conv_b = (const float*)d_in[3];   // [2048]
    const float* W_x    = (const float*)d_in[4];   // [2048,96]
    const float* W_dt   = (const float*)d_in[5];   // [64,2048]
    const float* b_dt   = (const float*)d_in[6];   // [2048]
    const float* A_log  = (const float*)d_in[7];   // [2048,16]
    const float* Dvec   = (const float*)d_in[8];   // [2048]
    const float* W_out  = (const float*)d_in[9];   // [2048,1024]
    float* out = (float*)d_out;                    // [2,2048,1024]

    char* ws = (char*)d_ws;
    size_t off = 0;
    auto carve = [&](size_t bytes) -> char* {
        char* p = ws + off;
        off = align256(off + bytes);
        return p;
    };

    bf16_t* x_bf    = (bf16_t*)carve((size_t)M_TOK * D_MODEL * 2);
    bf16_t* Wt_in   = (bf16_t*)carve((size_t)(2 * D_INNER) * D_MODEL * 2);
    float*  xz      = (float*) carve((size_t)M_TOK * 2 * D_INNER * 4);
    float*  u_f32   = (float*) carve((size_t)M_TOK * D_INNER * 4);
    bf16_t* u_bf    = (bf16_t*)carve((size_t)M_TOK * D_INNER * 2);
    bf16_t* Wt_x    = (bf16_t*)carve((size_t)XDBL_LD * D_INNER * 2);   // padded to 128 rows
    float*  x_dbl   = (float*) carve((size_t)M_TOK * XDBL_LD * 4);     // padded cols
    bf16_t* dt_bf   = (bf16_t*)carve((size_t)M_TOK * DT_RANK * 2);
    bf16_t* Wt_dt   = (bf16_t*)carve((size_t)D_INNER * DT_RANK * 2);
    float*  predel  = (float*) carve((size_t)M_TOK * D_INNER * 4);
    float*  ys      = (float*) carve((size_t)M_TOK * D_INNER * 4);
    bf16_t* y_bf    = (bf16_t*)carve((size_t)M_TOK * D_INNER * 2);
    bf16_t* Wt_out  = (bf16_t*)carve((size_t)D_MODEL * D_INNER * 2);
    (void)ws_size; (void)in_sizes; (void)n_in; (void)out_size;

    const int T256 = 256;
    auto blks = [](long long n, int t) { return (unsigned)((n + t - 1) / t); };

    // --- precision prep: convert x, transpose+convert all weights ---------
    k_convert_bf16  <<<blks((long long)M_TOK * D_MODEL, T256), T256, 0, stream>>>(x, x_bf, M_TOK * D_MODEL);
    k_transpose_bf16<<<blks((long long)D_MODEL * 2 * D_INNER, T256), T256, 0, stream>>>(W_in,  Wt_in,  D_MODEL, 2 * D_INNER);
    k_transpose_bf16<<<blks((long long)D_INNER * (DT_RANK + 2 * D_STATE), T256), T256, 0, stream>>>(W_x, Wt_x, D_INNER, DT_RANK + 2 * D_STATE);
    k_transpose_bf16<<<blks((long long)DT_RANK * D_INNER, T256), T256, 0, stream>>>(W_dt,  Wt_dt,  DT_RANK, D_INNER);
    k_transpose_bf16<<<blks((long long)D_INNER * D_MODEL, T256), T256, 0, stream>>>(W_out, Wt_out, D_INNER, D_MODEL);

    // --- GEMM 1: xz = x @ W_in  [4096 x 4096], K=1024 ---------------------
    k_wmma_gemm<M_TOK, 2 * D_INNER, D_MODEL>
        <<<dim3((2 * D_INNER) / BN, M_TOK / BM), 256, 0, stream>>>(x_bf, Wt_in, xz);

    // --- conv + silu -> u -------------------------------------------------
    k_conv_silu<<<blks((long long)M_TOK * D_INNER, T256), T256, 0, stream>>>(xz, conv_w, conv_b, u_f32, u_bf);

    // --- GEMM 2: x_dbl = u @ W_x  [4096 x 128(pad 96)], K=2048 ------------
    k_wmma_gemm<M_TOK, XDBL_LD, D_INNER>
        <<<dim3(XDBL_LD / BN, M_TOK / BM), 256, 0, stream>>>(u_bf, Wt_x, x_dbl);

    // --- GEMM 3: pre_delta = dt @ W_dt  [4096 x 2048], K=64 ---------------
    k_slice_dt<<<blks((long long)M_TOK * DT_RANK, T256), T256, 0, stream>>>(x_dbl, dt_bf);
    k_wmma_gemm<M_TOK, D_INNER, DT_RANK>
        <<<dim3(D_INNER / BN, M_TOK / BM), 256, 0, stream>>>(dt_bf, Wt_dt, predel);

    // --- selective scan ---------------------------------------------------
    k_scan<<<blks((long long)B_SZ * D_INNER, T256), T256, 0, stream>>>(predel, b_dt, x_dbl, u_f32, A_log, ys);

    // --- gate -------------------------------------------------------------
    k_gate<<<blks((long long)M_TOK * D_INNER, T256), T256, 0, stream>>>(ys, u_f32, Dvec, xz, y_bf);

    // --- GEMM 4: out = y @ W_out  [4096 x 1024], K=2048 -> d_out ----------
    k_wmma_gemm<M_TOK, D_MODEL, D_INNER>
        <<<dim3(D_MODEL / BN, M_TOK / BM), 256, 0, stream>>>(y_bf, Wt_out, out);
}
